// EncoderLayer_56745107915517
// MI455X (gfx1250) — matrix operand order repaired and checked by reading, not compiled, not run on hardware
//
#include <hip/hip_runtime.h>
#include <hip/hip_bf16.h>
#include <cstdint>
#include <cstddef>

#define B_    4
#define T_    4096
#define D_    512
#define H_    2048
#define TOPK  16
#define KWIN  25
#define PAD_  12
#define NROWS (B_ * T_)

typedef __attribute__((ext_vector_type(16))) __bf16        bf16x16;
typedef __attribute__((ext_vector_type(8)))  float         floatx8;
typedef __attribute__((ext_vector_type(4)))  unsigned int  uint32x4;

union BfVec { bf16x16 v; uint32x4 q[2]; };

__device__ __forceinline__ unsigned short f2bf(float f) {
    unsigned int u = __float_as_uint(f);
    unsigned int r = u + 0x7FFFu + ((u >> 16) & 1u);   // round-to-nearest-even
    return (unsigned short)(r >> 16);
}

// ---- CDNA5 async copy: global -> LDS (ASYNCcnt tracked), inline asm per ISA 10.x
__device__ __forceinline__ void async_load_16B(unsigned lds_off, const void* gaddr) {
    asm volatile("global_load_async_to_lds_b128 %0, %1, off"
                 :: "v"(lds_off), "v"(gaddr) : "memory");
}
__device__ __forceinline__ void wait_async0() {          // all async copies done
    asm volatile("s_wait_asynccnt 0x0" ::: "memory");
}
__device__ __forceinline__ void wait_async1() {          // all but newest done
    asm volatile("s_wait_asynccnt 0x1" ::: "memory");
}

// B-matrix 32x16 bf16 tile from a transposed [N,K] row-major matrix:
// B[K,N] = BT[N,K]. lanes<16: K=k0+0..7,16..23, lanes>=16: K=k0+8..15,24..31, N = lane&15.
__device__ __forceinline__ bf16x16 load_bt_tile(const unsigned short* p, int ld,
                                                int n0, int k0, int lane) {
    int n  = lane & 15;
    int kb = (lane < 16) ? 0 : 8;
    const unsigned short* r = p + (size_t)(n0 + n) * ld + k0 + kb;
    BfVec x;
    x.q[0] = *(const uint32x4*)(r);
    x.q[1] = *(const uint32x4*)(r + 16);
    return x.v;
}

__device__ __forceinline__ floatx8 wmma_bf16(bf16x16 a, bf16x16 b, floatx8 c) {
    return __builtin_amdgcn_wmma_f32_16x16x32_bf16(false, a, false, b, (short)0, c,
                                                   false, false);
}

// -------- shared WMMA mainloop: A (row-major [M,KT]) staged via async-to-LDS
// double buffer (16x32 bf16 tile shared by all 8 waves of the block); each wave
// computes NT 16x16 output tiles (one A fragment reused NT times).
// Pipeline: issue prefetch(buf^1) then wait ASYNCcnt<=1 -> tile[buf] resident
// (in-order completion) while the new copy stays in flight across the WMMAs.
template<int KT, int NT>
__device__ __forceinline__ void gemm_mainloop(const unsigned short* __restrict__ A, int m0,
                                              const unsigned short* __restrict__ BT, int n0,
                                              unsigned short (&tileA)[2][16][32],
                                              floatx8 (&c)[NT], int tid, int lane) {
    int srow = tid >> 2, sch = tid & 3;           // stager: 64 threads x 16B
    if (tid < 64) {
        async_load_16B((unsigned)(uintptr_t)&tileA[0][srow][sch * 8],
                       A + (size_t)(m0 + srow) * KT + sch * 8);
    }
    int buf = 0;
    for (int kk = 0; kk < KT; kk += 32) {
        if (kk + 32 < KT) {                       // uniform branch
            if (tid < 64) {
                async_load_16B((unsigned)(uintptr_t)&tileA[buf ^ 1][srow][sch * 8],
                               A + (size_t)(m0 + srow) * KT + kk + 32 + sch * 8);
            }
            wait_async1();                        // tile[buf] done, prefetch in flight
        } else {
            wait_async0();                        // last tile: ensure copy complete
        }
        __syncthreads();                          // publish tile[buf] to all waves

        int m  = lane & 15;
        int kb = (lane < 16) ? 0 : 8;
        const unsigned short* rp = &tileA[buf][m][kb];
        BfVec a;
        a.q[0] = *(const uint32x4*)(rp);
        a.q[1] = *(const uint32x4*)(rp + 16);
        #pragma unroll
        for (int t = 0; t < NT; ++t) {
            bf16x16 b = load_bt_tile(BT, KT, n0 + t * 16, kk, lane);
            c[t] = wmma_bf16(a.v, b, c[t]);
        }
        __syncthreads();                          // reads of tile[buf] retired
        buf ^= 1;
    }
}

// ---------------------------------------------------------------- utilities
__global__ void __launch_bounds__(256) cvt_bf16_kernel(const float* in,
                                                       unsigned short* out, int n) {
    int i = blockIdx.x * 256 + threadIdx.x;
    if (i < n) out[i] = f2bf(in[i]);
}

// out[c*rows + r] = bf16(in[r*cols + c])  (produce [N,K] transposed bf16 weights)
__global__ void __launch_bounds__(256) transpose_bf16_kernel(const float* in,
                                                             unsigned short* out,
                                                             int rows, int cols) {
    int i = blockIdx.x * 256 + threadIdx.x;
    if (i < rows * cols) {
        int r = i / cols, c = i - r * cols;
        out[(size_t)c * rows + r] = f2bf(in[i]);
    }
}

__global__ void __launch_bounds__(256) zero_f32_kernel(float* p, int n) {
    int i = blockIdx.x * 256 + threadIdx.x;
    if (i < n) p[i] = 0.0f;
}

// ------------------------------------------------- autocorrelation via WMMA Gram
// mean_corr[b,tau] numerator = sum of Gram elements G[i,j]=<x_i,x_j> with
// (i-j) mod T == tau.  Each wave computes one 16x16 WMMA tile of G and folds it
// into 31 diagonal bins in LDS (ds_add_f32), then 31 global atomics.
__global__ void __launch_bounds__(256) gram_autocorr_kernel(const unsigned short* xb,
                                                            float* corr) {
    __shared__ __align__(16) unsigned short tileA[2][16][32];
    __shared__ float bins[8][32];
    int tid  = threadIdx.x;
    int lane = tid & 31;
    int w    = tid >> 5;
    int ti   = blockIdx.x;               // row tile (i / 16) -- shared A tile
    int tj   = blockIdx.y * 8 + w;       // col tile (j / 16) -- per wave
    int b    = blockIdx.z;
    const unsigned short* xbat = xb + (size_t)b * T_ * D_;

    bins[w][lane] = 0.0f;
    floatx8 c[1] = {};
    gemm_mainloop<D_, 1>(xbat, ti * 16, xbat, tj * 16, tileA, c, tid, lane);

    int mb = (lane < 16) ? 0 : 8;        // C layout: M = r + mb, N = lane&15
    int n  = lane & 15;
    #pragma unroll
    for (int r = 0; r < 8; ++r) {
        int idx = (r + mb) - n + 15;     // 0..30
        atomicAdd(&bins[w][idx], c[0][r]);
    }
    __syncthreads();

    if (lane < 31) {
        float v = bins[w][lane];
        int tau = (ti - tj) * 16 + (lane - 15);
        tau = ((tau % T_) + T_) % T_;
        atomicAdd(&corr[b * T_ + tau], v);
    }
}

// -------------------------------------------------------- top-k + softmax weights
__global__ void __launch_bounds__(256) topk_softmax_kernel(const float* corr,
                                                           int* delays, float* weights) {
    __shared__ float vals[T_];
    __shared__ float sval[256];
    __shared__ int   sidx[256];
    __shared__ float topv[TOPK];
    int b = blockIdx.x, tid = threadIdx.x;
    for (int i = tid; i < T_; i += 256) vals[i] = corr[b * T_ + i] * (1.0f / D_);
    __syncthreads();

    for (int it = 0; it < TOPK; ++it) {
        float best = -3.4e38f; int bi = 0;
        for (int i = tid; i < T_; i += 256) {
            float v = vals[i];
            if (v > best) { best = v; bi = i; }
        }
        sval[tid] = best; sidx[tid] = bi;
        __syncthreads();
        for (int s = 128; s > 0; s >>= 1) {
            if (tid < s && sval[tid + s] > sval[tid]) {
                sval[tid] = sval[tid + s]; sidx[tid] = sidx[tid + s];
            }
            __syncthreads();
        }
        if (tid == 0) {
            topv[it] = sval[0];
            delays[b * TOPK + it] = sidx[0];
            vals[sidx[0]] = -3.4e38f;
        }
        __syncthreads();
    }
    if (tid == 0) {
        float mx = topv[0];                 // descending order -> [0] is max
        float e[TOPK], s = 0.0f;
        #pragma unroll
        for (int i = 0; i < TOPK; ++i) { e[i] = __expf(topv[i] - mx); s += e[i]; }
        float inv = 1.0f / s;
        #pragma unroll
        for (int i = 0; i < TOPK; ++i) weights[b * TOPK + i] = e[i] * inv;
    }
}

// ------------------------------------- delay-gather + residual + LayerNorm (h)
__global__ void __launch_bounds__(256) gather_ln1_kernel(const float* x,
                                                         const int* delays,
                                                         const float* weights,
                                                         const float* g1, const float* be1,
                                                         float* h) {
    int row = blockIdx.x;                 // b*T + t
    int b = row >> 12, t = row & (T_ - 1);
    int tid = threadIdx.x;
    __shared__ int   dly[TOPK];
    __shared__ float wgt[TOPK];
    __shared__ float red[256];
    if (tid < TOPK) { dly[tid] = delays[b * TOPK + tid]; wgt[tid] = weights[b * TOPK + tid]; }
    __syncthreads();

    const float* xb = x + (size_t)b * T_ * D_;
    float y[2];
    #pragma unroll
    for (int j = 0; j < 2; ++j) {
        int d = tid + j * 256;
        float acc = xb[(size_t)t * D_ + d];
        #pragma unroll
        for (int k = 0; k < TOPK; ++k) {
            int tt = (t + dly[k]) & (T_ - 1);     // roll(-delay): v[(t+delay)%T]
            acc += wgt[k] * xb[(size_t)tt * D_ + d];
        }
        y[j] = acc;
    }

    red[tid] = y[0] + y[1];
    __syncthreads();
    for (int s = 128; s > 0; s >>= 1) { if (tid < s) red[tid] += red[tid + s]; __syncthreads(); }
    float mean = red[0] * (1.0f / D_);
    __syncthreads();
    float v0 = y[0] - mean, v1 = y[1] - mean;
    red[tid] = v0 * v0 + v1 * v1;
    __syncthreads();
    for (int s = 128; s > 0; s >>= 1) { if (tid < s) red[tid] += red[tid + s]; __syncthreads(); }
    float rstd = rsqrtf(red[0] * (1.0f / D_) + 1e-5f);

    float* hr = h + (size_t)row * D_;
    #pragma unroll
    for (int j = 0; j < 2; ++j) {
        int d = tid + j * 256;
        hr[d] = (y[j] - mean) * rstd * g1[d] + be1[d];
    }
}

// --------------------------- series decomposition: 25-tap edge-replicated MA
__global__ void __launch_bounds__(256) decomp_kernel(const float* in, float* seas,
                                                     float* trend, unsigned short* seas_bf) {
    int row = blockIdx.x;
    int b = row >> 12, t = row & (T_ - 1);
    int tid = threadIdx.x;
    const float* base = in + (size_t)b * T_ * D_;
    #pragma unroll
    for (int j = 0; j < 2; ++j) {
        int d = tid + j * 256;
        float s = 0.0f;
        #pragma unroll
        for (int o = -PAD_; o <= PAD_; ++o) {
            int tt = t + o;
            tt = tt < 0 ? 0 : (tt > T_ - 1 ? T_ - 1 : tt);
            s += base[(size_t)tt * D_ + d];
        }
        float tr = s * (1.0f / KWIN);
        float se = base[(size_t)t * D_ + d] - tr;
        size_t oi = (size_t)row * D_ + d;
        seas[oi] = se;
        if (trend)   trend[oi]   = tr;
        if (seas_bf) seas_bf[oi] = f2bf(se);
    }
}

// ----------------------------------------- GEMM1: relu(seas @ w1 + b1) -> bf16
// block: 16 rows x 512 cols (8 waves x 16x64 strip); K = D_ = 512
__global__ void __launch_bounds__(256) gemm1_relu_kernel(const unsigned short* A,
                                                         const unsigned short* BT,
                                                         const float* bias,
                                                         unsigned short* out) {
    __shared__ __align__(16) unsigned short tileA[2][16][32];
    int tid = threadIdx.x, lane = tid & 31, w = tid >> 5;
    int m0 = blockIdx.x * 16;
    int n0 = blockIdx.y * 512 + w * 64;
    floatx8 c[4] = {};
    gemm_mainloop<D_, 4>(A, m0, BT, n0, tileA, c, tid, lane);

    int mb = (lane < 16) ? 0 : 8;
    #pragma unroll
    for (int t = 0; t < 4; ++t) {
        int n = n0 + t * 16 + (lane & 15);
        float bv = bias[n];
        #pragma unroll
        for (int r = 0; r < 8; ++r) {
            float v = c[t][r] + bv;
            v = v > 0.0f ? v : 0.0f;
            out[(size_t)(m0 + mb + r) * H_ + n] = f2bf(v);
        }
    }
}

// ----------------------------------------- GEMM2: hid @ w2 + b2 -> fp32
// block: 16 rows x 512 cols (8 waves x 16x64 strip); K = H_ = 2048
__global__ void __launch_bounds__(256) gemm2_bias_kernel(const unsigned short* A,
                                                         const unsigned short* BT,
                                                         const float* bias, float* out) {
    __shared__ __align__(16) unsigned short tileA[2][16][32];
    int tid = threadIdx.x, lane = tid & 31, w = tid >> 5;
    int m0 = blockIdx.x * 16;
    int n0 = w * 64;                       // N = 512 covered by one block row
    floatx8 c[4] = {};
    gemm_mainloop<H_, 4>(A, m0, BT, n0, tileA, c, tid, lane);

    int mb = (lane < 16) ? 0 : 8;
    #pragma unroll
    for (int t = 0; t < 4; ++t) {
        int n = n0 + t * 16 + (lane & 15);
        float bv = bias[n];
        #pragma unroll
        for (int r = 0; r < 8; ++r)
            out[(size_t)(m0 + mb + r) * D_ + n] = c[t][r] + bv;
    }
}

// ----------------------------------------- residual + LayerNorm 2
__global__ void __launch_bounds__(256) add_ln_kernel(const float* a, const float* bsrc,
                                                     const float* g, const float* be,
                                                     float* out) {
    int row = blockIdx.x, tid = threadIdx.x;
    __shared__ float red[256];
    const float* ar = a + (size_t)row * D_;
    const float* br = bsrc + (size_t)row * D_;
    float y[2];
    #pragma unroll
    for (int j = 0; j < 2; ++j) { int d = tid + j * 256; y[j] = ar[d] + br[d]; }

    red[tid] = y[0] + y[1];
    __syncthreads();
    for (int s = 128; s > 0; s >>= 1) { if (tid < s) red[tid] += red[tid + s]; __syncthreads(); }
    float mean = red[0] * (1.0f / D_);
    __syncthreads();
    float v0 = y[0] - mean, v1 = y[1] - mean;
    red[tid] = v0 * v0 + v1 * v1;
    __syncthreads();
    for (int s = 128; s > 0; s >>= 1) { if (tid < s) red[tid] += red[tid + s]; __syncthreads(); }
    float rstd = rsqrtf(red[0] * (1.0f / D_) + 1e-5f);

    float* orow = out + (size_t)row * D_;
    #pragma unroll
    for (int j = 0; j < 2; ++j) {
        int d = tid + j * 256;
        orow[d] = (y[j] - mean) * rstd * g[d] + be[d];
    }
}

// =============================================================== launcher
extern "C" void kernel_launch(void* const* d_in, const int* in_sizes, int n_in,
                              void* d_out, int out_size, void* d_ws, size_t ws_size,
                              hipStream_t stream) {
    const float* x   = (const float*)d_in[0];
    const float* w1  = (const float*)d_in[1];
    const float* b1  = (const float*)d_in[2];
    const float* w2  = (const float*)d_in[3];
    const float* b2  = (const float*)d_in[4];
    const float* g1  = (const float*)d_in[5];
    const float* be1 = (const float*)d_in[6];
    const float* g2  = (const float*)d_in[7];
    const float* be2 = (const float*)d_in[8];

    char* ws = (char*)d_ws;
    size_t off = 0;
    auto alloc = [&](size_t bytes) -> char* {
        char* p = ws + off;
        off = (off + bytes + 255) & ~(size_t)255;
        return p;
    };
    const size_t NE = (size_t)B_ * T_ * D_;           // 8.4M elems
    unsigned short* xb      = (unsigned short*)alloc(NE * 2);
    unsigned short* w1T     = (unsigned short*)alloc((size_t)D_ * H_ * 2);
    unsigned short* w2T     = (unsigned short*)alloc((size_t)H_ * D_ * 2);
    float*          corr    = (float*)alloc((size_t)B_ * T_ * 4);
    int*            delays  = (int*)alloc((size_t)B_ * TOPK * 4);
    float*          weights = (float*)alloc((size_t)B_ * TOPK * 4);
    float*          h       = (float*)alloc(NE * 4);
    float*          seas    = (float*)alloc(NE * 4);
    unsigned short* seas_bf = (unsigned short*)alloc(NE * 2);
    unsigned short* hid     = (unsigned short*)alloc((size_t)NROWS * H_ * 2);
    float*          ff      = (float*)alloc(NE * 4);
    float*          ff2     = h;                       // reuse: h dead after decomp1

    float* seas_out  = (float*)d_out;
    float* trend_out = seas_out + NE;

    // 1. x -> bf16
    cvt_bf16_kernel<<<(int)((NE + 255) / 256), 256, 0, stream>>>(x, xb, (int)NE);
    // 2. weight transposes to [N,K] bf16
    transpose_bf16_kernel<<<(D_ * H_ + 255) / 256, 256, 0, stream>>>(w1, w1T, D_, H_);
    transpose_bf16_kernel<<<(H_ * D_ + 255) / 256, 256, 0, stream>>>(w2, w2T, H_, D_);
    // 3. zero corr accumulator
    zero_f32_kernel<<<(B_ * T_ + 255) / 256, 256, 0, stream>>>(corr, B_ * T_);
    // 4. Gram-tile autocorrelation (WMMA + async-staged A + LDS binning)
    gram_autocorr_kernel<<<dim3(T_ / 16, T_ / 16 / 8, B_), 256, 0, stream>>>(xb, corr);
    // 5. top-16 + softmax
    topk_softmax_kernel<<<B_, 256, 0, stream>>>(corr, delays, weights);
    // 6. delay-gather + residual + LN1
    gather_ln1_kernel<<<NROWS, 256, 0, stream>>>(x, delays, weights, g1, be1, h);
    // 7. decomposition 1 (seasonal fp32 + bf16 for GEMM)
    decomp_kernel<<<NROWS, 256, 0, stream>>>(h, seas, nullptr, seas_bf);
    // 8. GEMM1 + bias + ReLU (WMMA, async-staged A)
    gemm1_relu_kernel<<<dim3(NROWS / 16, H_ / 512), 256, 0, stream>>>(seas_bf, w1T, b1, hid);
    // 9. GEMM2 + bias (WMMA, async-staged A)
    gemm2_bias_kernel<<<dim3(NROWS / 16, 1), 256, 0, stream>>>(hid, w2T, b2, ff);
    // 10. residual + LN2
    add_ln_kernel<<<NROWS, 256, 0, stream>>>(seas, ff, g2, be2, ff2);
    // 11. final decomposition -> (seasonal, trend) into d_out
    decomp_kernel<<<NROWS, 256, 0, stream>>>(ff2, seas_out, trend_out, nullptr);

    (void)in_sizes; (void)n_in; (void)out_size; (void)ws_size;
}
